// AdditiveAttention_76991583748113
// MI455X (gfx1250) — matrix operand order repaired and checked by reading, not compiled, not run on hardware
//
#include <hip/hip_runtime.h>
#include <hip/hip_fp16.h>

typedef __attribute__((ext_vector_type(16))) _Float16 v16h;
typedef __attribute__((ext_vector_type(8)))  float    v8f;

#define B_  16
#define Q_  128
#define K_  512
#define QS_ 256
#define H_  128
#define DV_ 256

#define QT_ 8     // query rows per score block
#define CK_ 64    // keys per async LDS chunk (64 rows * 512B = 32KB)

__device__ __forceinline__ float fast_tanh(float x) {
#if __has_builtin(__builtin_amdgcn_tanhf)
    return __builtin_amdgcn_tanhf(x);   // gfx1250 hardware v_tanh_f32
#else
    return tanhf(x);
#endif
}

// ---------------------------------------------------------------------------
// Prep A: W_q / W_k [QS,H] f32  ->  WqT / WkT [H,QS] f16  (tiny, 64K elems)
// ---------------------------------------------------------------------------
__global__ __launch_bounds__(256)
void wt_kernel(const float* __restrict__ Wq, const float* __restrict__ Wk,
               _Float16* __restrict__ WqT, _Float16* __restrict__ WkT) {
    const int id    = blockIdx.x * 256 + threadIdx.x;     // 0..65535
    const int which = (id >= QS_ * H_) ? 1 : 0;
    const int r     = id - which * QS_ * H_;
    const int h     = r >> 8;                             // / QS_
    const int k     = r & (QS_ - 1);
    const float*  W  = which ? Wk  : Wq;
    _Float16*     WT = which ? WkT : WqT;
    WT[h * QS_ + k] = (_Float16)W[k * H_ + h];
}

// ---------------------------------------------------------------------------
// Prep B: values [B,K,DV] f32 -> valT [B,DV,K] f16, LDS 32x32 tiled transpose
// ---------------------------------------------------------------------------
__global__ __launch_bounds__(256)
void transpose_values_kernel(const float* __restrict__ values, _Float16* __restrict__ valT) {
    const int blk = blockIdx.x;                 // B * (DV/32=8) * (K/32=16) = 2048
    const int b   = blk >> 7;
    const int d0  = ((blk >> 4) & 7) * 32;
    const int k0  = (blk & 15) * 32;
    __shared__ _Float16 tile[32][33];
    const int tr = threadIdx.x >> 5;            // 0..7
    const int tc = threadIdx.x & 31;
    const float* src = values + (size_t)b * K_ * DV_;
#pragma unroll
    for (int i = 0; i < 4; ++i)
        tile[tr + i * 8][tc] = (_Float16)src[(size_t)(k0 + tr + i * 8) * DV_ + d0 + tc];
    __syncthreads();
    _Float16* dst = valT + (size_t)b * DV_ * K_;
#pragma unroll
    for (int i = 0; i < 4; ++i)
        dst[(size_t)(d0 + tr + i * 8) * K_ + k0 + tc] = tile[tc][tr + i * 8];
}

// ---------------------------------------------------------------------------
// Kernel 1: fused projections q = queries @ W_q, k = keys @ W_k (WMMA f16).
// One wave per 16x16 f32 tile; B fragments are table-order f16 columns of WT.
// ---------------------------------------------------------------------------
__global__ __launch_bounds__(128)
void proj_kernel(const float* __restrict__ queries, const float* __restrict__ keys,
                 const _Float16* __restrict__ WqT,  const _Float16* __restrict__ WkT,
                 float* __restrict__ qproj,         float* __restrict__ kproj) {
    const int lane = threadIdx.x & 31;
    const int wave = threadIdx.x >> 5;
    const int tile = blockIdx.x * 4 + wave;      // 5120 tiles
    const int row_tile = tile >> 3;              // 640 row tiles
    const int col_tile = tile & 7;               // 8 col tiles (H=128)
    const int hg = lane >> 4;
    const int lm = lane & 15;

    const int  row0 = row_tile * 16;
    const bool is_q = (row0 < B_ * Q_);
    const float*    src = is_q ? queries : keys;
    const _Float16* WT  = is_q ? WqT     : WkT;
    float*          dst = is_q ? qproj   : kproj;
    const int       rb  = is_q ? row0    : row0 - B_ * Q_;

    const float*    arow = src + (size_t)(rb + lm) * QS_;
    const int       col  = col_tile * 16 + lm;
    const _Float16* wcol = WT + (size_t)col * QS_;

    v8f acc = {};
    for (int kb = 0; kb < QS_; kb += 32) {
        v16h af, bf;
        const int ka0 = kb + hg * 8;
        const int ka1 = kb + 16 + hg * 8;
        const int kb0 = kb + hg * 16;
#pragma unroll
        for (int e = 0; e < 8; ++e) {
            af[e]     = (_Float16)arow[ka0 + e];
            af[e + 8] = (_Float16)arow[ka1 + e];
        }
#pragma unroll
        for (int e = 0; e < 16; ++e) bf[e] = wcol[kb + hg * 8 + (e & 7) + ((e >> 3) << 4)];   // table-order f16
        acc = __builtin_amdgcn_wmma_f32_16x16x32_f16(false, af, false, bf,
                                                     (short)0, acc, false, false);
    }
#pragma unroll
    for (int v = 0; v < 8; ++v)
        dst[(size_t)(rb + hg * 8 + v) * H_ + col] = acc[v];
}

// ---------------------------------------------------------------------------
// Kernel 2: scores + mask + softmax. One block = (b, 8 query rows).
// k-rows double-buffered into LDS via global_load_async_to_lds_b128;
// hardware tanh; 4-lane shuffle reduce per key; per-wave softmax.
// ---------------------------------------------------------------------------
__global__ __launch_bounds__(256)
void score_softmax_kernel(const float* __restrict__ qproj, const float* __restrict__ kproj,
                          const float* __restrict__ wv,    const int* __restrict__ valid_lens,
                          _Float16* __restrict__ attn) {
    const int blk = blockIdx.x;                 // B * (Q/QT) = 16*16 = 256
    const int b   = blk >> 4;
    const int q0  = (blk & 15) * QT_;
    const int t   = threadIdx.x;

    __shared__ float qs[QT_][H_];               // 4 KB
    __shared__ float wvs[H_];
    __shared__ float kbuf[2][CK_ * H_];         // 2 x 32 KB
    __shared__ float scores[QT_][K_];           // 16 KB

    for (int i = t; i < QT_ * H_; i += 256)
        qs[i >> 7][i & (H_ - 1)] = qproj[(size_t)(b * Q_ + q0) * H_ + i];
    if (t < H_) wvs[t] = wv[t];
    for (int i = t; i < QT_ * K_; i += 256) (&scores[0][0])[i] = -1e6f;

    const int valid = valid_lens[b];
    const int nch   = (valid + CK_ - 1) / CK_;  // 1..8
    const unsigned long long kgbase = (unsigned long long)(size_t)(kproj + (size_t)b * K_ * H_);

    auto issue = [&](int c) {
        const int lanebyte = t * 16;                         // 0..4095
#pragma unroll
        for (int j = 0; j < 8; ++j) {
            const int off = j * 4096 + lanebyte;             // byte offset in 32KB chunk
            const unsigned lds = (unsigned)(size_t)(&kbuf[c & 1][0]) + off;
            const unsigned long long g = kgbase + (unsigned long long)c * (CK_ * H_ * 4) + off;
            asm volatile("global_load_async_to_lds_b128 %0, %1, off"
                         :: "v"(lds), "v"(g) : "memory");
        }
    };

    issue(0);
    for (int c = 0; c < nch; ++c) {
        if (c + 1 < nch) {
            issue(c + 1);
            asm volatile("s_wait_asynccnt 0x8" ::: "memory");  // chunk c complete
        } else {
            asm volatile("s_wait_asynccnt 0x0" ::: "memory");
        }
        __syncthreads();                                       // chunk c visible block-wide

        const int kk = c * CK_ + (t >> 2);                     // key owned by 4 lanes
        const int hb = (t & 3) * 32;                           // h-slice
        const float* krow = &kbuf[c & 1][(t >> 2) * H_ + hb];
        float acc[QT_] = {};
        for (int h = 0; h < 32; h += 4) {
            const float4 kv = *(const float4*)(krow + h);
            const int hh = hb + h;
#pragma unroll
            for (int qq = 0; qq < QT_; ++qq) {
                acc[qq] += wvs[hh + 0] * fast_tanh(qs[qq][hh + 0] + kv.x);
                acc[qq] += wvs[hh + 1] * fast_tanh(qs[qq][hh + 1] + kv.y);
                acc[qq] += wvs[hh + 2] * fast_tanh(qs[qq][hh + 2] + kv.z);
                acc[qq] += wvs[hh + 3] * fast_tanh(qs[qq][hh + 3] + kv.w);
            }
        }
#pragma unroll
        for (int qq = 0; qq < QT_; ++qq) {                     // combine 4 h-slices
            acc[qq] += __shfl_xor(acc[qq], 1, 32);
            acc[qq] += __shfl_xor(acc[qq], 2, 32);
        }
        if (kk < valid) {
            const int j = t & 3;                               // lane j writes rows 2j,2j+1
            scores[2 * j][kk]     = acc[2 * j];
            scores[2 * j + 1][kk] = acc[2 * j + 1];
        }
        __syncthreads();                                       // kbuf reuse + scores ready
    }

    // softmax: wave w owns query row w (8 waves, 32 lanes x 16 entries)
    const int w = t >> 5, lane = t & 31;
    float vals[16];
    float rmax = -1e30f;
#pragma unroll
    for (int i = 0; i < 16; ++i) { vals[i] = scores[w][lane + i * 32]; rmax = fmaxf(rmax, vals[i]); }
    for (int m = 16; m; m >>= 1) rmax = fmaxf(rmax, __shfl_xor(rmax, m, 32));
    float rsum = 0.f;
#pragma unroll
    for (int i = 0; i < 16; ++i) { vals[i] = __expf(vals[i] - rmax); rsum += vals[i]; }
    for (int m = 16; m; m >>= 1) rsum += __shfl_xor(rsum, m, 32);
    const float inv = 1.0f / rsum;

    _Float16* arow = attn + (size_t)(b * Q_ + q0 + w) * K_;
#pragma unroll
    for (int i = 0; i < 16; ++i) arow[lane + i * 32] = (_Float16)(vals[i] * inv);
}

// ---------------------------------------------------------------------------
// Kernel 3: out = attn @ values (WMMA f16), B from pre-transposed valT f16.
// k-loop clamped to ceil(valid/32) since attn is exactly 0 beyond valid_len.
// ---------------------------------------------------------------------------
__global__ __launch_bounds__(128)
void out_kernel(const _Float16* __restrict__ attn, const _Float16* __restrict__ valT,
                const int* __restrict__ valid_lens, float* __restrict__ out) {
    const int lane = threadIdx.x & 31;
    const int wave = threadIdx.x >> 5;
    const int tile = blockIdx.x * 4 + wave;      // 2048 tiles
    const int dt = tile & 15;
    const int qt = (tile >> 4) & 7;
    const int b  = tile >> 7;
    const int hg = lane >> 4;
    const int lm = lane & 15;

    const int valid = valid_lens[b];
    const int kmax  = min(K_, (valid + 31) & ~31);

    const _Float16* arow = attn + (size_t)(b * Q_ + qt * 16 + lm) * K_;
    const int       col  = dt * 16 + lm;
    const _Float16* vcol = valT + ((size_t)b * DV_ + col) * K_;

    v8f acc = {};
    for (int kb = 0; kb < kmax; kb += 32) {
        v16h af, bf;
        const int ka0 = kb + hg * 8;
        const int ka1 = kb + 16 + hg * 8;
        const int kb0 = kb + hg * 16;
#pragma unroll
        for (int e = 0; e < 8; ++e) {
            af[e]     = arow[ka0 + e];
            af[e + 8] = arow[ka1 + e];
        }
#pragma unroll
        for (int e = 0; e < 16; ++e) bf[e] = vcol[kb + hg * 8 + (e & 7) + ((e >> 3) << 4)];    // table-order f16
        acc = __builtin_amdgcn_wmma_f32_16x16x32_f16(false, af, false, bf,
                                                     (short)0, acc, false, false);
    }
#pragma unroll
    for (int v = 0; v < 8; ++v)
        out[(size_t)(b * Q_ + qt * 16 + hg * 8 + v) * DV_ + col] = acc[v];
}

extern "C" void kernel_launch(void* const* d_in, const int* in_sizes, int n_in,
                              void* d_out, int out_size, void* d_ws, size_t ws_size,
                              hipStream_t stream) {
    const float* queries = (const float*)d_in[0];
    const float* keys    = (const float*)d_in[1];
    const float* values  = (const float*)d_in[2];
    const int*   vlens   = (const int*)  d_in[3];
    const float* Wq      = (const float*)d_in[4];
    const float* Wk      = (const float*)d_in[5];
    const float* wv      = (const float*)d_in[6];
    float*       out     = (float*)d_out;

    char* ws = (char*)d_ws;
    size_t off = 0;
    float*    qproj = (float*)(ws + off);    off += (size_t)B_ * Q_ * H_ * 4;   // 1 MB
    float*    kproj = (float*)(ws + off);    off += (size_t)B_ * K_ * H_ * 4;   // 4 MB
    _Float16* attn  = (_Float16*)(ws + off); off += (size_t)B_ * Q_ * K_ * 2;   // 2 MB
    _Float16* WqT   = (_Float16*)(ws + off); off += (size_t)QS_ * H_ * 2;       // 64 KB
    _Float16* WkT   = (_Float16*)(ws + off); off += (size_t)QS_ * H_ * 2;       // 64 KB
    _Float16* valT  = (_Float16*)(ws + off); off += (size_t)B_ * K_ * DV_ * 2;  // 4 MB

    wt_kernel<<<256, 256, 0, stream>>>(Wq, Wk, WqT, WkT);
    transpose_values_kernel<<<2048, 256, 0, stream>>>(values, valT);
    proj_kernel<<<1280, 128, 0, stream>>>(queries, keys, WqT, WkT, qproj, kproj);
    score_softmax_kernel<<<256, 256, 0, stream>>>(qproj, kproj, wv, vlens, attn);
    out_kernel<<<512, 128, 0, stream>>>(attn, valT, vlens, out);
}
